// MesaNetLayer_15075335209567
// MI455X (gfx1250) — matrix operand order repaired and checked by reading, not compiled, not run on hardware
//
#include <hip/hip_runtime.h>
#include <cstdint>

// ---------------------------------------------------------------------------
// MesaNet layer for MI455X (gfx1250, wave32).
// Big GEMMs: bf16 V_WMMA_F32_16X16X32_BF16, f32 accumulation.
// Block tile 64 x BN (128 or 64), 256 threads = 8 waves, wave tile 32 x BN/4 (32x32 or 32x16).
// B operand is pre-transposed to [N][K] so both A and B tiles stage into LDS
// with b128 stores only; LDS is double-buffered (1 barrier / K-step) and
// next-tile global loads overlap the WMMA chain.
// ---------------------------------------------------------------------------

typedef __bf16 bf16;
typedef __attribute__((ext_vector_type(16))) __bf16 v16bf;
typedef __attribute__((ext_vector_type(8)))  __bf16 v8bf;
typedef __attribute__((ext_vector_type(8)))  float  v8f;

#define BB 8
#define SS 4096
#define DD 1024
#define AA 64
#define HH 8
#define BS 32768  // B*S

// ---------------------------------------------------------------------------
// Batched bf16 GEMM: C[z] = act( A[z] * B[z]^T? ... B given K-major ) + bias.
// A: row-major [M][K] (lda). BT: K-major [N][K] (ldbt). K multiple of 32.
// M = gridDim.y*64, N = gridDim.x*BN. Writes f32 and/or bf16 output.
// ---------------------------------------------------------------------------
template <int BN>
__global__ __launch_bounds__(256) void gemm_bf16_kernel(
    const bf16* __restrict__ A, int lda, long sA,
    const bf16* __restrict__ BT, int ldbt, long sB,
    const float* __restrict__ bias,
    float* __restrict__ Cf, bf16* __restrict__ Cb, int ldc, long sC,
    int K, int relu)
{
  constexpr int NSUB = BN / 64;              // n-subtiles per wave (1 or 2)
  __shared__ bf16 lA[2][64][40];             // 64 x 32 (+8 pad), double buffered
  __shared__ bf16 lBT[2][BN][40];            // BN x 32 (+8 pad), K-contiguous

  const int tid  = threadIdx.x;
  const int w    = tid >> 5;
  const int lane = tid & 31;
  const int g    = lane >> 4;                // half-wave group
  const int lr   = lane & 15;
  const int wm   = (w >> 2) * 32;            // wave row offset in tile
  const int wn   = (w & 3) * (BN / 4);       // wave col offset in tile

  const long m_base = (long)blockIdx.y * 64;
  const long n_base = (long)blockIdx.x * BN;
  const long zA = sA * (long)blockIdx.z;
  const long zB = sB * (long)blockIdx.z;
  const long zC = sC * (long)blockIdx.z;

  v8f acc[2][NSUB];
#pragma unroll
  for (int mi = 0; mi < 2; ++mi)
#pragma unroll
    for (int ni = 0; ni < NSUB; ++ni)
#pragma unroll
      for (int i = 0; i < 8; ++i) acc[mi][ni][i] = 0.0f;

  // A staging: thread -> (row, 8-half chunk)
  const int ar = tid >> 2;                   // 0..63
  const int ac = (tid & 3) * 8;              // 0,8,16,24
  // B staging: NSUB chunks/thread -> (n row, 8-half chunk)
  int bn[NSUB], bc[NSUB];
#pragma unroll
  for (int p = 0; p < NSUB; ++p) {
    const int idx = tid + p * 256;
    bn[p] = idx >> 2;                        // 0..BN-1
    bc[p] = (idx & 3) * 8;
  }

  const bf16* Ap  = A  + zA + (m_base + ar) * (long)lda + ac;
  const bf16* BTp0 = BT + zB + (n_base + bn[0]) * (long)ldbt + bc[0];
  const bf16* BTp1 = (NSUB > 1) ? (BT + zB + (n_base + bn[NSUB - 1]) * (long)ldbt + bc[NSUB - 1])
                                : BTp0;

  // prologue: stage tile 0 into buffer 0
  uint4 aReg = *(const uint4*)(Ap);
  uint4 bReg[NSUB];
  bReg[0] = *(const uint4*)(BTp0);
  if (NSUB > 1) bReg[NSUB - 1] = *(const uint4*)(BTp1);
  *(uint4*)&lA[0][ar][ac] = aReg;
#pragma unroll
  for (int p = 0; p < NSUB; ++p) *(uint4*)&lBT[0][bn[p]][bc[p]] = bReg[p];

  int cur = 0;
  for (int k0 = 0; k0 < K; k0 += 32) {
    __syncthreads();                         // current buffer visible
    const bool hasNext = (k0 + 32) < K;
    if (hasNext) {                           // overlap next-tile loads with WMMA
      aReg = *(const uint4*)(Ap + k0 + 32);
      bReg[0] = *(const uint4*)(BTp0 + k0 + 32);
      if (NSUB > 1) bReg[NSUB - 1] = *(const uint4*)(BTp1 + k0 + 32);
      if (k0 + 64 < K) {                     // global_prefetch_b8 one tile ahead
        __builtin_prefetch(Ap + k0 + 64, 0, 1);
        __builtin_prefetch(BTp0 + k0 + 64, 0, 1);
      }
    }

    // ---- fragments per ISA 7.12.2 layouts ----
    v16bf af[2];
#pragma unroll
    for (int mi = 0; mi < 2; ++mi) {
      const int row = wm + mi * 16 + lr;
      v8bf lo = *(const v8bf*)&lA[cur][row][8 * g];
      v8bf hi = *(const v8bf*)&lA[cur][row][16 + 8 * g];
#pragma unroll
      for (int i = 0; i < 8; ++i) { af[mi][i] = lo[i]; af[mi][i + 8] = hi[i]; }
    }
    v16bf bfr[NSUB];
#pragma unroll
    for (int ni = 0; ni < NSUB; ++ni) {
      const int col = wn + ni * 16 + lr;
      v8bf lo = *(const v8bf*)&lBT[cur][col][8 * g];
      v8bf hi = *(const v8bf*)&lBT[cur][col][16 + 8 * g];
#pragma unroll
      for (int i = 0; i < 8; ++i) { bfr[ni][i] = lo[i]; bfr[ni][i + 8] = hi[i]; }
    }

#pragma unroll
    for (int mi = 0; mi < 2; ++mi)
#pragma unroll
      for (int ni = 0; ni < NSUB; ++ni)
        acc[mi][ni] = __builtin_amdgcn_wmma_f32_16x16x32_bf16(
            false, af[mi], false, bfr[ni], (short)0, acc[mi][ni], false, false);

    if (hasNext) {                           // stage next tile into other buffer
      const int nxt = cur ^ 1;
      *(uint4*)&lA[nxt][ar][ac] = aReg;
#pragma unroll
      for (int p = 0; p < NSUB; ++p) *(uint4*)&lBT[nxt][bn[p]][bc[p]] = bReg[p];
    }
    cur ^= 1;
  }

  // ---- epilogue: bias + optional ReLU, f32/bf16 stores ----
#pragma unroll
  for (int mi = 0; mi < 2; ++mi)
#pragma unroll
    for (int ni = 0; ni < NSUB; ++ni) {
      const long col = n_base + wn + ni * 16 + lr;
      const float bv = bias ? bias[col] : 0.0f;
#pragma unroll
      for (int r = 0; r < 8; ++r) {
        const long rowm = m_base + wm + mi * 16 + r + 8 * g;
        float v = acc[mi][ni][r] + bv;
        if (relu) v = fmaxf(v, 0.0f);
        const long off = zC + rowm * (long)ldc + col;
        if (Cf) Cf[off] = v;
        if (Cb) Cb[off] = (bf16)v;
      }
    }
}

// ---------------------------------------------------------------------------
// Tiled transposes (32x32 tiles through LDS): out[C][R] = in[R][C], batched.
// ---------------------------------------------------------------------------
__global__ __launch_bounds__(256) void k_transpose_f32_bf16(
    const float* __restrict__ in, bf16* __restrict__ out,
    int R, int C, long sIn, long sOut) {
  __shared__ bf16 t[32][33];
  const int tx = threadIdx.x & 31, ty = threadIdx.x >> 5;
  const long zi = sIn * (long)blockIdx.z, zo = sOut * (long)blockIdx.z;
  const int c0 = blockIdx.x * 32, r0 = blockIdx.y * 32;
#pragma unroll
  for (int i = 0; i < 32; i += 8)
    t[ty + i][tx] = (bf16)in[zi + (long)(r0 + ty + i) * C + c0 + tx];
  __syncthreads();
#pragma unroll
  for (int i = 0; i < 32; i += 8)
    out[zo + (long)(c0 + ty + i) * R + r0 + tx] = t[tx][ty + i];
}

__global__ __launch_bounds__(256) void k_transpose_bf16(
    const bf16* __restrict__ in, bf16* __restrict__ out,
    int R, int C, long sIn, long sOut) {
  __shared__ bf16 t[32][33];
  const int tx = threadIdx.x & 31, ty = threadIdx.x >> 5;
  const long zi = sIn * (long)blockIdx.z, zo = sOut * (long)blockIdx.z;
  const int c0 = blockIdx.x * 32, r0 = blockIdx.y * 32;
#pragma unroll
  for (int i = 0; i < 32; i += 8)
    t[ty + i][tx] = in[zi + (long)(r0 + ty + i) * C + c0 + tx];
  __syncthreads();
#pragma unroll
  for (int i = 0; i < 32; i += 8)
    out[zo + (long)(c0 + ty + i) * R + r0 + tx] = t[tx][ty + i];
}

// combined[r, 0:1024] = x[r,:]; combined[r, 1024+a] = efas[r]*ep_w[a] + ep_b[a]
__global__ void k_build_combined(const float* __restrict__ x, const float* __restrict__ efas,
                                 const float* __restrict__ epw, const float* __restrict__ epb,
                                 bf16* __restrict__ comb, long total) {
  long i = (long)blockIdx.x * blockDim.x + threadIdx.x;
  if (i >= total) return;
  long row = i / 1088;
  int  col = (int)(i - row * 1088);
  float v;
  if (col < DD) v = x[row * DD + col];
  else { int a = col - DD; v = efas[row] * epw[a] + epb[a]; }
  comb[i] = (bf16)v;
}

// per-row softmax over A=64; emits rw (row-major bf16) and rwT[b][a][s] bf16
__global__ void k_softmax_router(const float* __restrict__ logits,
                                 bf16* __restrict__ rw, bf16* __restrict__ rwT) {
  long r = (long)blockIdx.x * blockDim.x + threadIdx.x;
  if (r >= BS) return;
  const float* lp = logits + r * AA;
  float mx = -1e30f;
  for (int i = 0; i < AA; ++i) mx = fmaxf(mx, lp[i]);
  float e[AA]; float sum = 0.0f;
  for (int i = 0; i < AA; ++i) { e[i] = __expf(lp[i] - mx); sum += e[i]; }
  const float inv = 1.0f / sum;
  const long b = r >> 12, s = r & 4095;
  for (int i = 0; i < AA; ++i) {
    const float wv = e[i] * inv;
    rw[r * AA + i] = (bf16)wv;
    rwT[(b * AA + i) * SS + s] = (bf16)wv;
  }
}

// per-(b,h) attention over 64 anchors, head dim 128; ctx written bf16
__global__ __launch_bounds__(64) void k_attention(const float* __restrict__ qkv,
                                                  bf16* __restrict__ ctx) {
  const int bb = blockIdx.x >> 3, h = blockIdx.x & 7;
  const int t = threadIdx.x;                 // 0..63
  __shared__ float lk[64][128];
  __shared__ float at[64][64];
  const long rowbase = (long)bb * AA;
  const float* kp = qkv + (rowbase + t) * 3072 + 1024 + h * 128;
  for (int d = 0; d < 128; ++d) lk[t][d] = kp[d];
  __syncthreads();
  const float scale = 0.08838834764831845f;  // 1/sqrt(128)
  for (int a = 0; a < 64; ++a) {
    const float* qp = qkv + (rowbase + a) * 3072 + h * 128;
    float acc = 0.0f;
    for (int d = 0; d < 128; ++d) acc += qp[d] * lk[t][d];
    at[a][t] = acc * scale;
  }
  __syncthreads();
  float mx = -1e30f;
  for (int j = 0; j < 64; ++j) mx = fmaxf(mx, at[t][j]);
  float sum = 0.0f;
  for (int j = 0; j < 64; ++j) { float e = __expf(at[t][j] - mx); at[t][j] = e; sum += e; }
  const float inv = 1.0f / sum;
  for (int j = 0; j < 64; ++j) at[t][j] *= inv;
  __syncthreads();
  bf16* cp = ctx + (rowbase + t) * DD + h * 128;
  for (int d = 0; d < 128; ++d) {
    float acc = 0.0f;
    for (int j = 0; j < 64; ++j)
      acc += at[t][j] * qkv[(rowbase + j) * 3072 + 2048 + h * 128 + d];
    cp[d] = (bf16)acc;
  }
}

// out = LayerNorm(a + b) * g + beta ; writes f32 and/or bf16 (null to skip)
__global__ __launch_bounds__(256) void k_ln(const float* __restrict__ a,
                                            const float* __restrict__ b,
                                            const float* __restrict__ gw,
                                            const float* __restrict__ bw,
                                            float* __restrict__ outf,
                                            bf16* __restrict__ outb) {
  const long row = blockIdx.x;
  const int t = threadIdx.x;
  __shared__ float red[256];
  float v[4];
  float s = 0.0f;
#pragma unroll
  for (int i = 0; i < 4; ++i) {
    const int c = t + i * 256;
    const float val = a[row * DD + c] + b[row * DD + c];
    v[i] = val; s += val;
  }
  red[t] = s; __syncthreads();
  for (int off = 128; off; off >>= 1) { if (t < off) red[t] += red[t + off]; __syncthreads(); }
  const float mean = red[0] * (1.0f / DD);
  __syncthreads();
  float s2 = 0.0f;
#pragma unroll
  for (int i = 0; i < 4; ++i) { const float d = v[i] - mean; s2 += d * d; }
  red[t] = s2; __syncthreads();
  for (int off = 128; off; off >>= 1) { if (t < off) red[t] += red[t + off]; __syncthreads(); }
  const float inv = rsqrtf(red[0] * (1.0f / DD) + 1e-5f);
#pragma unroll
  for (int i = 0; i < 4; ++i) {
    const int c = t + i * 256;
    const float y = (v[i] - mean) * inv * gw[c] + bw[c];
    if (outf) outf[row * DD + c] = y;
    if (outb) outb[row * DD + c] = (bf16)y;
  }
}

// ---------------------------------------------------------------------------
// Host orchestration
// ---------------------------------------------------------------------------
extern "C" void kernel_launch(void* const* d_in, const int* in_sizes, int n_in,
                              void* d_out, int out_size, void* d_ws, size_t ws_size,
                              hipStream_t stream) {
  (void)in_sizes; (void)n_in; (void)out_size; (void)ws_size;
  const float* x      = (const float*)d_in[0];
  const float* efas   = (const float*)d_in[1];
  const float* ep_w   = (const float*)d_in[2];
  const float* ep_b   = (const float*)d_in[3];
  const float* rm_w1  = (const float*)d_in[5];
  const float* rm_b1  = (const float*)d_in[6];
  const float* rm_w2  = (const float*)d_in[7];
  const float* rm_b2  = (const float*)d_in[8];
  const float* in_w   = (const float*)d_in[9];
  const float* in_b   = (const float*)d_in[10];
  const float* out_w  = (const float*)d_in[11];
  const float* out_b  = (const float*)d_in[12];
  const float* f1w    = (const float*)d_in[13];
  const float* f1b    = (const float*)d_in[14];
  const float* f2w    = (const float*)d_in[15];
  const float* f2b    = (const float*)d_in[16];
  const float* ln1g   = (const float*)d_in[17];
  const float* ln1b   = (const float*)d_in[18];
  const float* ln2g   = (const float*)d_in[19];
  const float* ln2b   = (const float*)d_in[20];

  // bump allocator over workspace (256B aligned)
  size_t o = 0;
  auto alloc = [&](size_t bytes) -> char* {
    size_t a = (o + 255) & ~(size_t)255;
    o = a + bytes;
    return (char*)d_ws + a;
  };
  // transposed bf16 weights [N][K]
  bf16*  wrm1T   = (bf16*)alloc((size_t)1024 * 1088 * 2);
  bf16*  wrm2T   = (bf16*)alloc((size_t)64 * 1024 * 2);
  bf16*  w_inT   = (bf16*)alloc((size_t)3072 * 1024 * 2);
  bf16*  w_outT  = (bf16*)alloc((size_t)1024 * 1024 * 2);
  bf16*  wf1T    = (bf16*)alloc((size_t)4096 * 1024 * 2);
  bf16*  wf2T    = (bf16*)alloc((size_t)1024 * 4096 * 2);
  // activations
  bf16*  comb    = (bf16*)alloc((size_t)BS * 1088 * 2);
  bf16*  xT      = (bf16*)alloc((size_t)BB * DD * SS * 2);   // [b][d][s]
  bf16*  hidden  = (bf16*)alloc((size_t)BS * 1024 * 2);
  float* logits  = (float*)alloc((size_t)BS * 64 * 4);
  bf16*  rw      = (bf16*)alloc((size_t)BS * 64 * 2);
  bf16*  rwT     = (bf16*)alloc((size_t)BB * 64 * SS * 2);   // [b][a][s]
  bf16*  anch    = (bf16*)alloc((size_t)BB * 64 * 1024 * 2);
  float* qkv     = (float*)alloc((size_t)BB * 64 * 3072 * 4);
  bf16*  ctx     = (bf16*)alloc((size_t)BB * 64 * 1024 * 2);
  bf16*  mixed   = (bf16*)alloc((size_t)BB * 64 * 1024 * 2);
  bf16*  mixedT  = (bf16*)alloc((size_t)BB * 1024 * 64 * 2); // [b][d][a]
  float* tu      = (float*)alloc((size_t)BS * 1024 * 4);
  float* h       = (float*)alloc((size_t)BS * 1024 * 4);
  bf16*  hb      = (bf16*)comb;              // alias: comb dead after router GEMM1
  bf16*  ffn1    = (bf16*)alloc((size_t)BS * 4096 * 2);
  float* ffn2    = tu;                       // alias: tu dead after LN1

  auto cdiv = [](long a, long b) { return (unsigned)((a + b - 1) / b); };

  // 1) weights -> bf16, transposed to [N][K]
  k_transpose_f32_bf16<<<dim3(1024/32, 1088/32, 1), 256, 0, stream>>>(rm_w1, wrm1T, 1088, 1024, 0, 0);
  k_transpose_f32_bf16<<<dim3(64/32,   1024/32, 1), 256, 0, stream>>>(rm_w2, wrm2T, 1024, 64,   0, 0);
  k_transpose_f32_bf16<<<dim3(3072/32, 1024/32, 1), 256, 0, stream>>>(in_w,  w_inT, 1024, 3072, 0, 0);
  k_transpose_f32_bf16<<<dim3(1024/32, 1024/32, 1), 256, 0, stream>>>(out_w, w_outT,1024, 1024, 0, 0);
  k_transpose_f32_bf16<<<dim3(4096/32, 1024/32, 1), 256, 0, stream>>>(f1w,   wf1T,  1024, 4096, 0, 0);
  k_transpose_f32_bf16<<<dim3(1024/32, 4096/32, 1), 256, 0, stream>>>(f2w,   wf2T,  4096, 1024, 0, 0);
  // x -> xT[b][d][s] bf16 (B operand of anchors GEMM)
  k_transpose_f32_bf16<<<dim3(1024/32, 4096/32, BB), 256, 0, stream>>>(
      x, xT, SS, DD, (long)SS * DD, (long)SS * DD);

  // 2) combined = [x | energy] bf16
  k_build_combined<<<cdiv((long)BS*1088, 256), 256, 0, stream>>>(
      x, efas, ep_w, ep_b, comb, (long)BS * 1088);

  // 3) hidden = relu(combined @ rm_w1 + b1)   [32768,1088]x[1088,1024]
  gemm_bf16_kernel<128><<<dim3(1024/128, BS/64, 1), 256, 0, stream>>>(
      comb, 1088, 0, wrm1T, 1088, 0, rm_b1, nullptr, hidden, 1024, 0, 1088, 1);

  // 4) logits = hidden @ rm_w2 + b2           [32768,1024]x[1024,64]
  gemm_bf16_kernel<64><<<dim3(1, BS/64, 1), 256, 0, stream>>>(
      hidden, 1024, 0, wrm2T, 1024, 0, rm_b2, logits, nullptr, 64, 0, 1024, 0);

  // 5) routing softmax -> rw, rwT
  k_softmax_router<<<cdiv(BS, 256), 256, 0, stream>>>(logits, rw, rwT);

  // 6) anchors[b] = rwT[b] @ x[b]   [64,4096]x[4096,1024] per batch
  gemm_bf16_kernel<128><<<dim3(1024/128, 1, BB), 256, 0, stream>>>(
      rwT, SS, (long)64 * SS, xT, SS, (long)DD * SS, nullptr,
      nullptr, anch, 1024, (long)64 * 1024, SS, 0);

  // 7) qkv = anchors @ mha_in_w + b           [512,1024]x[1024,3072]
  gemm_bf16_kernel<128><<<dim3(3072/128, (BB*64)/64, 1), 256, 0, stream>>>(
      anch, 1024, 0, w_inT, 1024, 0, in_b, qkv, nullptr, 3072, 0, 1024, 0);

  // 8) attention per (b,h)
  k_attention<<<BB * HH, 64, 0, stream>>>(qkv, ctx);

  // 9) mixed = ctx @ mha_out_w + b            [512,1024]x[1024,1024]
  gemm_bf16_kernel<128><<<dim3(1024/128, (BB*64)/64, 1), 256, 0, stream>>>(
      ctx, 1024, 0, w_outT, 1024, 0, out_b, nullptr, mixed, 1024, 0, 1024, 0);

  // 9b) mixed -> mixedT[b][d][a]
  k_transpose_bf16<<<dim3(1024/32, 64/32, BB), 256, 0, stream>>>(
      mixed, mixedT, 64, 1024, (long)64 * 1024, (long)64 * 1024);

  // 10) token_updates[b] = rw[b] @ mixed[b]   [4096,64]x[64,1024] per batch
  gemm_bf16_kernel<128><<<dim3(1024/128, SS/64, BB), 256, 0, stream>>>(
      rw, 64, (long)SS * 64, mixedT, 64, (long)64 * 1024, nullptr,
      tu, nullptr, 1024, (long)SS * 1024, 64, 0);

  // 11) h = LN1(x + tu)  (f32 + bf16 copies)
  k_ln<<<BS, 256, 0, stream>>>(x, tu, ln1g, ln1b, h, hb);

  // 12) ffn1 = relu(h @ ffn_w1 + b1)          [32768,1024]x[1024,4096]
  gemm_bf16_kernel<128><<<dim3(4096/128, BS/64, 1), 256, 0, stream>>>(
      hb, 1024, 0, wf1T, 1024, 0, f1b, nullptr, ffn1, 4096, 0, 1024, 1);

  // 13) ffn2 = ffn1 @ ffn_w2 + b2             [32768,4096]x[4096,1024]
  gemm_bf16_kernel<128><<<dim3(1024/128, BS/64, 1), 256, 0, stream>>>(
      ffn1, 4096, 0, wf2T, 4096, 0, f2b, ffn2, nullptr, 1024, 0, 4096, 0);

  // 14) out = LN2(h + ffn2)
  k_ln<<<BS, 256, 0, stream>>>(h, ffn2, ln2g, ln2b, (float*)d_out, nullptr);
}
